// PraxisAttention_31860067402138
// MI455X (gfx1250) — matrix operand order repaired and checked by reading, not compiled, not run on hardware
//
#include <hip/hip_runtime.h>
#include <hip/hip_bf16.h>
#include <stdint.h>

// ---------------------------------------------------------------------------
// Differential attention (DIFF-Transformer style) for MI455X / gfx1250.
// bf16 WMMA (v_wmma_f32_16x16x32_bf16) everywhere; flash-style softmax tiles.
// Round 4: pre-transposed bf16 weights + transposed V producer layout
//          -> ALL global->LDS staging is async DMA (ASYNCcnt), all LDS
//          fragment reads are contiguous ds_load_b128.
// ---------------------------------------------------------------------------

typedef __attribute__((ext_vector_type(16))) __bf16 v16bf;
typedef __attribute__((ext_vector_type(8)))  float  v8f;
typedef int v4i_ls __attribute__((vector_size(16)));   // param type of async builtin
using bf16 = __bf16;

#define B_  2
#define S_  2048
#define D_  1024
#define H_  16
#define HD_ 64

#if defined(__has_builtin)
#  if __has_builtin(__builtin_amdgcn_global_load_async_to_lds_b128)
#    define HAVE_ASYNC_LDS 1
#  endif
#  if __has_builtin(__builtin_amdgcn_s_wait_asynccnt)
#    define HAVE_WAIT_ASYNC 1
#  endif
#endif

#ifdef HAVE_ASYNC_LDS
// Async DMA: global -> LDS, 16B per lane, tracked by ASYNCcnt.
__device__ __forceinline__ void stage_b128(void* l, const void* g) {
    __builtin_amdgcn_global_load_async_to_lds_b128(
        (v4i_ls*)g, (v4i_ls*)l, 0, 0);
}
__device__ __forceinline__ void wait_async0() {
#ifdef HAVE_WAIT_ASYNC
    __builtin_amdgcn_s_wait_asynccnt(0);
#else
    asm volatile("s_wait_asynccnt 0x0" ::: "memory");
#endif
}
#else
__device__ __forceinline__ void stage_b128(void* l, const void* g) {
    *reinterpret_cast<uint4*>(l) = *reinterpret_cast<const uint4*>(g);
}
__device__ __forceinline__ void wait_async0() {}
#endif

__device__ __forceinline__ bf16 f2bf(float f) {
    uint32_t u = __float_as_uint(f);
    uint32_t r = (u + 0x7FFFu + ((u >> 16) & 1u)) >> 16;
    unsigned short s = (unsigned short)r;
    bf16 b;
    __builtin_memcpy(&b, &s, 2);
    return b;
}

__device__ __forceinline__ v8f wmma_bf16(v16bf a, v16bf b, v8f c) {
    return __builtin_amdgcn_wmma_f32_16x16x32_bf16(
        /*neg_a=*/false, a, /*neg_b=*/false, b,
        /*c_mod=*/(short)0, c, /*reuse_a=*/false, /*reuse_b=*/false);
}

// A-fragment: 16x32 bf16 region at `base` (row-major, leading dim ld).
// Lanes 0-15 hold M=lane, K={0..7,16..23}; lanes 16-31 K={8..15,24..31}.
// Both 8-element runs are contiguous -> 2x ds_load_b128 per lane.
__device__ __forceinline__ v16bf frag_a(const bf16* base, int ld, int lane) {
    int r  = lane & 15;
    int kb = (lane >> 4) * 8;
    const bf16* p = base + r * ld;
    v16bf a;
#pragma unroll
    for (int i = 0; i < 16; ++i) {
        int k = kb + i + ((i >= 8) ? 8 : 0);
        a[i] = p[k];
    }
    return a;
}

// B-fragment from K-contiguous (column-major / transposed-staged) source:
// B[k][n] = base[n*ld + k]. Lanes 0-15: N=lane, K={0..7,16..23}; lanes 16-31: K={8..15,24..31}.
// Two contiguous 8-element runs -> 2x ds_load_b128 per lane.
__device__ __forceinline__ v16bf frag_bT(const bf16* base, int ld, int lane) {
    int c  = lane & 15;
    int kb = (lane >> 4) * 8;
    const bf16* p = base + c * ld;
    v16bf b;
#pragma unroll
    for (int i = 0; i < 16; ++i) b[i] = p[kb + i + ((i >= 8) ? 8 : 0)];
    return b;
}

// ---------------------------------------------------------------------------
// f32 -> bf16 conversion (elementwise)
// ---------------------------------------------------------------------------
__global__ void cvt_f32_bf16(const float* __restrict__ in, bf16* __restrict__ out, int n) {
    int i = blockIdx.x * blockDim.x + threadIdx.x;
    if (i < n) out[i] = f2bf(in[i]);
}

// f32 (K x N, row-major) -> bf16 transposed (N x K).  K == N == 1024 here.
__global__ void cvt_tr_f32_bf16(const float* __restrict__ in, bf16* __restrict__ out,
                                int K, int N) {
    int i = blockIdx.x * blockDim.x + threadIdx.x;   // i = n*K + k
    int n = i / K, k = i - n * K;
    if (n < N) out[i] = f2bf(in[(size_t)k * N + n]);
}

// ---------------------------------------------------------------------------
// bf16 GEMM: C[M,N] = A[M,K] @ B[K,N], with B supplied TRANSPOSED (BT = N x K).
// Block tile 128x64, K-step 32 (== WMMA K). 8 waves, each 32x32.
// Fully async double-buffered LDS staging (A natural, BT natural).
// Store modes: 0 = f32 (m,n);  1 = bf16 (m,n);
//              2 = bf16 V-layout (b,h,hd,s)  [coalesced 16B per subtile]
// ---------------------------------------------------------------------------
__global__ __launch_bounds__(256) void gemm_bf16(
    const bf16* __restrict__ A, const bf16* __restrict__ BT,
    void* __restrict__ Cout, int M, int N, int Kd, int mode)
{
    __shared__ __align__(16) bf16 As [2][128][32];   // (m, k)
    __shared__ __align__(16) bf16 BsT[2][64][32];    // (n, k)

    const int tid = threadIdx.x, lane = tid & 31, w = tid >> 5;
    const int mBase = blockIdx.y * 128, nBase = blockIdx.x * 64;
    const int wm = (w & 3) * 32;   // wave row offset inside tile
    const int wn = (w >> 2) * 32;  // wave col offset inside tile

    auto stageA = [&](int k0, int buf) {
#pragma unroll
        for (int c = 0; c < 2; ++c) {
            int ch = c * 256 + tid;
            int row = ch >> 2, seg = (ch & 3) * 8;
            stage_b128(&As[buf][row][seg],
                       A + (size_t)(mBase + row) * Kd + k0 + seg);
        }
    };
    auto stageB = [&](int k0, int buf) {   // BT rows are K-contiguous -> pure DMA
        int row = tid >> 2, seg = (tid & 3) * 8;    // row = n (0..63)
        stage_b128(&BsT[buf][row][seg],
                   BT + (size_t)(nBase + row) * Kd + k0 + seg);
    };

    v8f acc[2][2] = {};

    stageA(0, 0);
    stageB(0, 0);
    wait_async0();
    __syncthreads();

    const int nk = Kd / 32;
    for (int kt = 0; kt < nk; ++kt) {
        const int buf = kt & 1;
        if (kt + 1 < nk) {                 // overlap next tile DMA with compute
            stageA((kt + 1) * 32, buf ^ 1);
            stageB((kt + 1) * 32, buf ^ 1);
        }
        if (kt + 2 < nk)                    // speculative L2 prefetch of k+2 slab
            __builtin_prefetch(A + (size_t)(mBase + (tid >> 1)) * Kd + (kt + 2) * 32, 0, 1);

        v16bf af0 = frag_a (&As [buf][wm     ][0], 32, lane);
        v16bf af1 = frag_a (&As [buf][wm + 16][0], 32, lane);
        v16bf bf0 = frag_bT(&BsT[buf][wn     ][0], 32, lane);
        v16bf bf1 = frag_bT(&BsT[buf][wn + 16][0], 32, lane);

        acc[0][0] = wmma_bf16(af0, bf0, acc[0][0]);
        acc[0][1] = wmma_bf16(af0, bf1, acc[0][1]);
        acc[1][0] = wmma_bf16(af1, bf0, acc[1][0]);
        acc[1][1] = wmma_bf16(af1, bf1, acc[1][1]);

        wait_async0();
        __syncthreads();
    }

    const int hi = (lane >> 4) * 8, nc = lane & 15;
#pragma unroll
    for (int i = 0; i < 2; ++i)
#pragma unroll
        for (int j = 0; j < 2; ++j) {
            if (mode == 2) {
                // V-layout: n -> (h, hd); 8 consecutive m (= b*S+s) per lane,
                // contiguous along s in the (b,h,hd,s) layout -> one 16B store.
                int m0 = mBase + wm + i * 16 + hi;
                int n  = nBase + wn + j * 16 + nc;
                int bb = m0 >> 11, s0 = m0 & (S_ - 1);
                int hh = n >> 6,  hd = n & (HD_ - 1);
                union { uint4 u; bf16 e[8]; } pk;
#pragma unroll
                for (int r = 0; r < 8; ++r) pk.e[r] = f2bf(acc[i][j][r]);
                bf16* dst = (bf16*)Cout +
                    (((size_t)bb * H_ + hh) * HD_ + hd) * S_ + s0;
                *reinterpret_cast<uint4*>(dst) = pk.u;
            } else {
#pragma unroll
                for (int r = 0; r < 8; ++r) {
                    int m = mBase + wm + i * 16 + hi + r;
                    int n = nBase + wn + j * 16 + nc;
                    if (mode == 1)
                        ((bf16*)Cout)[(size_t)m * N + n] = f2bf(acc[i][j][r]);
                    else
                        ((float*)Cout)[(size_t)m * N + n] = acc[i][j][r];
                }
            }
        }
}

// ---------------------------------------------------------------------------
// lambda[h] = 0.8 + exp(lq0.lk0) - exp(lq1.lk1)
// ---------------------------------------------------------------------------
__global__ void lambda_k(const float* __restrict__ lq, const float* __restrict__ lk,
                         float* __restrict__ lamb)
{
    int h = threadIdx.x;
    if (h >= H_) return;
    float d0 = 0.f, d1 = 0.f;
    for (int i = 0; i < HD_; ++i) {
        d0 += lq[h * HD_ + i] * lk[h * HD_ + i];
        d1 += lq[H_ * HD_ + h * HD_ + i] * lk[H_ * HD_ + h * HD_ + i];
    }
    lamb[h] = 0.8f + __expf(d0) - __expf(d1);
}

// ---------------------------------------------------------------------------
// Differential flash attention. One block = (b, h, 64-row q tile).
// Loops n in {0,1}; accumulates res = lam[h]*(O0 - O1) with online softmax.
// Q,K (n,b,s,D) bf16; V (b,h,hd,s) bf16 transposed. All tiles async-DMA'd.
// ---------------------------------------------------------------------------
__global__ __launch_bounds__(256) void diff_attn(
    const bf16* __restrict__ Qb, const bf16* __restrict__ Kb, const bf16* __restrict__ VT,
    const float* __restrict__ amask, const float* __restrict__ lamb,
    float* __restrict__ attn)
{
    __shared__ __align__(16) bf16  Qs [64][64];   // (q, hd)
    __shared__ __align__(16) bf16  Ks [64][64];   // (k, hd)
    __shared__ __align__(16) bf16  VsT[64][64];   // (hd, k)
    __shared__ __align__(16) float Ss [64][64];
    __shared__ __align__(16) bf16  Ps [64][64];
    __shared__ float mrow[64], lrow[64], arow[64];

    const int tid = threadIdx.x, lane = tid & 31, w = tid >> 5;
    const int b = blockIdx.z, h = blockIdx.y, qt = blockIdx.x;
    const int qb = (w & 3) * 16;   // wave's q-row block inside 64-row tile
    const int cb = (w >> 2) * 32;  // wave's col block base (2 x 16)
    const int hi = (lane >> 4), ncol = lane & 15;

    const float scale = 0.125f;                           // 1/sqrt(64)
    const float slope = exp2f(-0.5f * (float)(h + 1));    // 2^(-8(h+1)/16)
    const float lam   = lamb[h];

    float res[2][8] = {};
    const size_t n_stride = (size_t)B_ * S_ * D_;
    const bf16* Vhead = VT + ((size_t)b * H_ + h) * HD_ * S_;   // (hd, s)

    for (int n = 0; n < 2; ++n) {
        const bf16* Qn = Qb + (size_t)n * n_stride;
        const bf16* Kn = Kb + (size_t)n * n_stride;

        // Stage Q tile (async DMA, natural layout)
#pragma unroll
        for (int c = 0; c < 2; ++c) {
            int ch = c * 256 + tid;
            int row = ch >> 3, sg = (ch & 7) * 8;
            stage_b128(&Qs[row][sg],
                       Qn + ((size_t)(b * S_ + qt * 64 + row) * D_ + h * HD_ + sg));
        }
        if (tid < 64) { mrow[tid] = -1e30f; lrow[tid] = 0.f; }
        wait_async0();
        __syncthreads();

        v16bf aq0 = frag_a(&Qs[qb][0], 64, lane);
        v16bf aq1 = frag_a(&Qs[qb][32], 64, lane);
        v8f oacc[2] = {};

        const int nkt = qt + 1;  // causal: only k-tiles <= q-tile
        for (int kt = 0; kt < nkt; ++kt) {
            // K tile: async DMA, natural (k, hd) layout
            // V tile: async DMA from transposed (hd, s) layout
#pragma unroll
            for (int c = 0; c < 2; ++c) {
                int ch = c * 256 + tid;
                int row = ch >> 3, sg = (ch & 7) * 8;
                stage_b128(&Ks[row][sg],
                           Kn + ((size_t)(b * S_ + kt * 64 + row) * D_ + h * HD_ + sg));
                stage_b128(&VsT[row][sg],
                           Vhead + (size_t)row * S_ + kt * 64 + sg);
            }
            wait_async0();
            __syncthreads();

            // Scores: S = Q @ K^T * scale - slope*(q-k) + masks
#pragma unroll
            for (int t = 0; t < 2; ++t) {
                int kcb = cb + t * 16;
                v16bf bk0 = frag_bT(&Ks[kcb][0], 64, lane);   // B[k=hd][n=kpos]
                v16bf bk1 = frag_bT(&Ks[kcb][32], 64, lane);
                v8f c = {};
                c = wmma_bf16(aq0, bk0, c);
                c = wmma_bf16(aq1, bk1, c);

                int kabs = kt * 64 + kcb + ncol;
                float pm = (1.0f - amask[b * S_ + kabs]) * -1e9f;
#pragma unroll
                for (int r = 0; r < 8; ++r) {
                    int qrow = qb + hi * 8 + r;
                    int qabs = qt * 64 + qrow;
                    float s = c[r] * scale - slope * (float)(qabs - kabs) + pm;
                    if (kabs > qabs) s -= 1e9f;
                    Ss[qrow][kcb + ncol] = s;
                }
            }
            __syncthreads();

            // Online softmax bookkeeping: one thread per q row
            if (tid < 64) {
                float mo = mrow[tid], mx = mo;
                for (int j = 0; j < 64; ++j) mx = fmaxf(mx, Ss[tid][j]);
                float al = __expf(mo - mx);
                float sum = 0.f;
                for (int j = 0; j < 64; ++j) {
                    float p = __expf(Ss[tid][j] - mx);
                    sum += p;
                    Ps[tid][j] = f2bf(p);
                }
                mrow[tid] = mx;
                lrow[tid] = lrow[tid] * al + sum;
                arow[tid] = al;
            }
            __syncthreads();

            // O += P~ @ V (rescale accumulators by alpha first)
            v16bf ap0 = frag_a(&Ps[qb][0], 64, lane);
            v16bf ap1 = frag_a(&Ps[qb][32], 64, lane);
#pragma unroll
            for (int t = 0; t < 2; ++t) {
                int ocb = cb + t * 16;
                v16bf bv0 = frag_bT(&VsT[ocb][0], 64, lane);   // B[k=kpos][n=hd]
                v16bf bv1 = frag_bT(&VsT[ocb][32], 64, lane);
#pragma unroll
                for (int r = 0; r < 8; ++r) oacc[t][r] *= arow[qb + hi * 8 + r];
                oacc[t] = wmma_bf16(ap0, bv0, oacc[t]);
                oacc[t] = wmma_bf16(ap1, bv1, oacc[t]);
            }
            __syncthreads();
        }

        // Combine: res += sign_n * lam * O / l
        const float sgn = n ? -lam : lam;
#pragma unroll
        for (int t = 0; t < 2; ++t)
#pragma unroll
            for (int r = 0; r < 8; ++r) {
                float l = lrow[qb + hi * 8 + r];
                res[t][r] += sgn * oacc[t][r] / l;
            }
        __syncthreads();
    }

    // Write attn (b,h,s,hd) f32
#pragma unroll
    for (int t = 0; t < 2; ++t)
#pragma unroll
        for (int r = 0; r < 8; ++r) {
            int qrow = qb + hi * 8 + r;
            int hd = cb + t * 16 + ncol;
            attn[(((size_t)b * H_ + h) * S_ + qt * 64 + qrow) * HD_ + hd] = res[t][r];
        }
}

// ---------------------------------------------------------------------------
// GroupNorm stats per (b,h) over (s,hd):  stats = {mean, rstd}
// ---------------------------------------------------------------------------
__global__ __launch_bounds__(256) void gn_stats(const float* __restrict__ attn,
                                                float* __restrict__ stats)
{
    const int bh = blockIdx.x;
    const float* p = attn + (size_t)bh * S_ * HD_;
    const int tid = threadIdx.x;
    float s = 0.f, s2 = 0.f;
    for (int i = tid; i < S_ * HD_; i += 256) {
        float v = p[i];
        s += v; s2 += v * v;
    }
    __shared__ float rs[256], rs2[256];
    rs[tid] = s; rs2[tid] = s2;
    __syncthreads();
    for (int st = 128; st > 0; st >>= 1) {
        if (tid < st) { rs[tid] += rs[tid + st]; rs2[tid] += rs2[tid + st]; }
        __syncthreads();
    }
    if (tid == 0) {
        const float inv = 1.0f / (float)(S_ * HD_);
        float mean = rs[0] * inv;
        float var  = rs2[0] * inv - mean * mean;
        stats[bh * 2 + 0] = mean;
        stats[bh * 2 + 1] = rsqrtf(var + 1e-5f);
    }
}

// ---------------------------------------------------------------------------
// GroupNorm apply + relayout (b,h,s,hd) f32 -> (b,s,D) bf16
// ---------------------------------------------------------------------------
__global__ void gn_apply(const float* __restrict__ attn, const float* __restrict__ stats,
                         const float* __restrict__ gw, const float* __restrict__ gb,
                         bf16* __restrict__ out)
{
    int i = blockIdx.x * 256 + threadIdx.x;   // < B*H*S*HD
    int hd = i & (HD_ - 1);
    int s  = (i >> 6) & (S_ - 1);
    int h  = (i >> 17) & (H_ - 1);
    int b  = i >> 21;
    float mean = stats[(b * H_ + h) * 2 + 0];
    float rstd = stats[(b * H_ + h) * 2 + 1];
    float v = (attn[i] - mean) * rstd * gw[h * HD_ + hd] + gb[h * HD_ + hd];
    out[((size_t)(b * S_ + s)) * D_ + h * HD_ + hd] = f2bf(v);
}

// ---------------------------------------------------------------------------
// Host launcher
// ---------------------------------------------------------------------------
extern "C" void kernel_launch(void* const* d_in, const int* in_sizes, int n_in,
                              void* d_out, int out_size, void* d_ws, size_t ws_size,
                              hipStream_t stream)
{
    const float* X     = (const float*)d_in[0];
    const float* amask = (const float*)d_in[1];
    const float* qw    = (const float*)d_in[2];
    const float* kw    = (const float*)d_in[3];
    const float* vw    = (const float*)d_in[4];
    const float* ow    = (const float*)d_in[5];
    const float* lq    = (const float*)d_in[6];
    const float* lk    = (const float*)d_in[7];
    const float* gw    = (const float*)d_in[8];
    const float* gb    = (const float*)d_in[9];
    float* out = (float*)d_out;

    char* ws = (char*)d_ws;
    size_t off = 0;
    auto alloc = [&](size_t bytes) -> void* {
        void* p = ws + off;
        off += (bytes + 255) & ~(size_t)255;
        return p;
    };

    const size_t MS = (size_t)B_ * S_;        // 4096 rows
    bf16* Xb   = (bf16*)alloc(MS * D_ * 2);                 // 8 MB
    bf16* WqT  = (bf16*)alloc((size_t)2 * D_ * D_ * 2);     // 4 MB (transposed)
    bf16* WkT  = (bf16*)alloc((size_t)2 * D_ * D_ * 2);     // 4 MB (transposed)
    bf16* WvT  = (bf16*)alloc((size_t)D_ * D_ * 2);         // 2 MB (transposed)
    bf16* WoT  = (bf16*)alloc((size_t)D_ * D_ * 2);         // 2 MB (transposed)
    bf16* Qbuf = (bf16*)alloc((size_t)2 * MS * D_ * 2);     // 16 MB (n,b,s,D)
    bf16* Kbuf = (bf16*)alloc((size_t)2 * MS * D_ * 2);     // 16 MB (n,b,s,D)
    bf16* VT   = (bf16*)alloc(MS * D_ * 2);                 // 8 MB  (b,h,hd,s)
    float* attnb = (float*)alloc((size_t)B_ * H_ * S_ * HD_ * 4); // 16 MB
    bf16* Nb   = (bf16*)alloc(MS * D_ * 2);                 // 8 MB
    float* stats = (float*)alloc(B_ * H_ * 2 * 4);
    float* lamb  = (float*)alloc(H_ * 4);

    // ---- convert inputs to bf16 (weights transposed for K-contiguous DMA) ----
    const int DD = D_ * D_;
    cvt_f32_bf16<<<(int)(MS * D_) / 256, 256, 0, stream>>>(X, Xb, (int)(MS * D_));
    cvt_tr_f32_bf16<<<DD / 256, 256, 0, stream>>>(qw,      WqT,      D_, D_);
    cvt_tr_f32_bf16<<<DD / 256, 256, 0, stream>>>(qw + DD, WqT + DD, D_, D_);
    cvt_tr_f32_bf16<<<DD / 256, 256, 0, stream>>>(kw,      WkT,      D_, D_);
    cvt_tr_f32_bf16<<<DD / 256, 256, 0, stream>>>(kw + DD, WkT + DD, D_, D_);
    cvt_tr_f32_bf16<<<DD / 256, 256, 0, stream>>>(vw,      WvT,      D_, D_);
    cvt_tr_f32_bf16<<<DD / 256, 256, 0, stream>>>(ow,      WoT,      D_, D_);

    // ---- projections ----
    dim3 ggrid(D_ / 64, (unsigned)(MS / 128));
    const size_t pN = MS * D_;                // elems per n in Q/K buffers
    gemm_bf16<<<ggrid, 256, 0, stream>>>(Xb, WqT,      Qbuf,      (int)MS, D_, D_, 1);
    gemm_bf16<<<ggrid, 256, 0, stream>>>(Xb, WqT + DD, Qbuf + pN, (int)MS, D_, D_, 1);
    gemm_bf16<<<ggrid, 256, 0, stream>>>(Xb, WkT,      Kbuf,      (int)MS, D_, D_, 1);
    gemm_bf16<<<ggrid, 256, 0, stream>>>(Xb, WkT + DD, Kbuf + pN, (int)MS, D_, D_, 1);
    gemm_bf16<<<ggrid, 256, 0, stream>>>(Xb, WvT,      VT,        (int)MS, D_, D_, 2);

    // ---- lambda per head ----
    lambda_k<<<1, 32, 0, stream>>>(lq, lk, lamb);

    // ---- differential flash attention ----
    dim3 agrid(S_ / 64, H_, B_);
    diff_attn<<<agrid, 256, 0, stream>>>(Qbuf, Kbuf, VT, amask, lamb, attnb);

    // ---- group norm ----
    gn_stats<<<B_ * H_, 256, 0, stream>>>(attnb, stats);
    gn_apply<<<(B_ * H_ * S_ * HD_) / 256, 256, 0, stream>>>(attnb, stats, gw, gb, Nb);

    // ---- output projection (f32 out) ----
    gemm_bf16<<<ggrid, 256, 0, stream>>>(Nb, WoT, out, (int)MS, D_, D_, 0);

    (void)in_sizes; (void)n_in; (void)out_size; (void)ws_size;
}
